// TMA_ATT_53936199303688
// MI455X (gfx1250) — matrix operand order repaired and checked by reading, not compiled, not run on hardware
//
#include <hip/hip_runtime.h>
#include <stdint.h>

// ---- problem dims (compile-time) ----
#define B_   4
#define S_   2048
#define D_   1024
#define O_   512

typedef __attribute__((ext_vector_type(16))) __bf16   v16bf;
typedef __attribute__((ext_vector_type(8)))  float    v8f;
typedef __attribute__((ext_vector_type(4)))  uint32_t u32x4;
typedef __attribute__((ext_vector_type(4)))  float    f32x4;

union ABu { v16bf v; u32x4 q[2]; };

__device__ __forceinline__ unsigned short f32_to_bf16(float f) {
  unsigned int u = __float_as_uint(f);
  u += 0x7FFFu + ((u >> 16) & 1u);     // round-to-nearest-even
  return (unsigned short)(u >> 16);
}

__device__ __forceinline__ uint32_t pack_bf16(float lo, float hi) {
  return (uint32_t)f32_to_bf16(lo) | ((uint32_t)f32_to_bf16(hi) << 16);
}

// ---------------- vectorized cast: fp32 -> bf16, 8 elements/thread ----------------
__global__ __launch_bounds__(256) void cast_x_kernel(const float* __restrict__ in,
                                                     unsigned short* __restrict__ out,
                                                     int n8) {
  int i = blockIdx.x * blockDim.x + threadIdx.x;
  if (i < n8) {
    const f32x4* p = (const f32x4*)(in + (size_t)i * 8);
    f32x4 x0 = p[0], x1 = p[1];
    u32x4 v = {pack_bf16(x0[0], x0[1]), pack_bf16(x0[2], x0[3]),
               pack_bf16(x1[0], x1[1]), pack_bf16(x1[2], x1[3])};
    *(u32x4*)(out + (size_t)i * 8) = v;
  }
}

// ---------------- W [D,O] fp32 -> WT [O,D] bf16 ----------------
__global__ __launch_bounds__(256) void transpose_w_kernel(const float* __restrict__ W,
                                                          unsigned short* __restrict__ WT) {
  int i = blockIdx.x * blockDim.x + threadIdx.x;   // i = o*D + d
  if (i < O_ * D_) {
    int o = i / D_;
    int d = i - o * D_;
    WT[i] = f32_to_bf16(W[(size_t)d * O_ + o]);
  }
}

// ---------------- projection GEMM: Y[8192,512] = Xbf @ WT^T ----------------
// one wave computes a 16(M) x 64(N) tile; K loop over D in 32 steps of 32,
// software-pipelined depth 2; sched_barriers pin the prefetch ahead of compute.
__global__ __launch_bounds__(128) void proj_kernel(const unsigned short* __restrict__ X,
                                                   const unsigned short* __restrict__ WT,
                                                   unsigned short* __restrict__ Y,
                                                   int transposed) {
  const int lane = threadIdx.x & 31;
  const int wave = threadIdx.x >> 5;
  const int wid  = blockIdx.x * 4 + wave;   // 0..4095
  const int mt   = wid >> 3;                // 0..511
  const int ng   = wid & 7;                 // 64-wide N group
  const int m    = lane & 15;
  const int half = lane >> 4;

  v8f acc[4];
#pragma unroll
  for (int j = 0; j < 4; ++j) acc[j] = (v8f){0,0,0,0,0,0,0,0};

  const unsigned short* Arow = X + (size_t)(mt * 16 + m) * D_;
  const unsigned short* Wrow[4];
#pragma unroll
  for (int j = 0; j < 4; ++j)
    Wrow[j] = WT + (size_t)(ng * 64 + j * 16 + m) * D_;

  ABu a[2], bm[2][4];
  auto loadStage = [&](int d0, int s) {
    a[s].q[0] = *(const u32x4*)(Arow + d0 + half * 16);
    a[s].q[1] = *(const u32x4*)(Arow + d0 + half * 16 + 8);
#pragma unroll
    for (int j = 0; j < 4; ++j) {
      const u32x4* wp = (const u32x4*)(Wrow[j] + d0 + half * 16);
      bm[s][j].q[0] = wp[0];
      bm[s][j].q[1] = wp[1];
    }
  };

  loadStage(0, 0);
#pragma unroll
  for (int it = 0; it < 32; ++it) {
    const int cur = it & 1;
    if (it < 31) loadStage((it + 1) * 32, cur ^ 1);   // prefetch next K-slice
    __builtin_amdgcn_sched_barrier(0);                // keep prefetch above compute
#pragma unroll
    for (int j = 0; j < 4; ++j)
      acc[j] = __builtin_amdgcn_wmma_f32_16x16x32_bf16(
          false, a[cur].v, false, bm[cur][j].v, (short)0, acc[j], false, false);
    __builtin_amdgcn_sched_barrier(0);
  }

  if (!transposed) {
#pragma unroll
    for (int j = 0; j < 4; ++j) {
      const int o = ng * 64 + j * 16 + m;
#pragma unroll
      for (int r = 0; r < 8; ++r)
        Y[(size_t)(mt * 16 + half * 8 + r) * O_ + o] = f32_to_bf16(acc[j][r]);
    }
  } else {
    const int srow = mt * 16;
    const int b    = srow >> 11;
    const int s_in = srow & 2047;
#pragma unroll
    for (int j = 0; j < 4; ++j) {
      const int o = ng * 64 + j * 16 + m;
      u32x4 v = {pack_bf16(acc[j][0], acc[j][1]), pack_bf16(acc[j][2], acc[j][3]),
                 pack_bf16(acc[j][4], acc[j][5]), pack_bf16(acc[j][6], acc[j][7])};
      *(u32x4*)(Y + (size_t)b * O_ * S_ + (size_t)o * S_ + s_in + half * 8) = v;
    }
  }
}

// ---------------- fused flash attention ----------------
// 64 chunks of 32 keys: 32 QK wmma + one softmax + 16 PV wmma per chunk.
// Block = 4 waves = 2 q-tile groups; 2 waves share a Q tile in LDS and each
// owns a 256-wide output slice.
__global__ __launch_bounds__(128) void attn_kernel(const unsigned short* __restrict__ Q,
                                                   const unsigned short* __restrict__ K,
                                                   const unsigned short* __restrict__ VT,
                                                   float* __restrict__ Out) {
  __shared__ unsigned short Qs[2][16 * 512];   // 32 KB
  __shared__ unsigned short Ps[4][16 * 32];    //  4 KB

  const int lane  = threadIdx.x & 31;
  const int wave  = threadIdx.x >> 5;
  const int g     = wave >> 1;
  const int ohalf = wave & 1;
  const int qt    = blockIdx.x * 2 + g;     // 0..511
  const int b     = qt >> 7;
  const int q0    = (qt & 127) << 4;
  const int m     = lane & 15;
  const int half  = lane >> 4;

  // ---- async copy of the contiguous 16x512 bf16 Q tile into LDS ----
  // (ASYNCcnt-tracked global->LDS DMA path; no VGPR round trip)
  {
    const unsigned short* Qg = Q + (size_t)(b * S_ + q0) * O_;
    const int tid2 = ohalf * 32 + lane;     // 0..63 within the 2-wave group
    const uint32_t lds0 = (uint32_t)(uintptr_t)&Qs[g][0];
    const uint64_t gb0  = (uint64_t)(uintptr_t)Qg;
#pragma unroll
    for (int i = 0; i < 16; ++i) {
      uint32_t ldsa = lds0 + (uint32_t)((tid2 + i * 64) * 16);
      uint64_t ga   = gb0 + (uint64_t)((tid2 + i * 64) * 16);
      asm volatile("global_load_async_to_lds_b128 %0, %1, off"
                   :: "v"(ldsa), "v"(ga) : "memory");
    }
    asm volatile("s_wait_asynccnt 0x0" ::: "memory");
  }
  __syncthreads();

  v8f acc[16];
#pragma unroll
  for (int ot = 0; ot < 16; ++ot) acc[ot] = (v8f){0,0,0,0,0,0,0,0};
  float rmax[8], rsum[8];
#pragma unroll
  for (int r = 0; r < 8; ++r) { rmax[r] = -1e30f; rsum[r] = 0.0f; }

  const unsigned short* Kg = K + (size_t)b * S_ * O_;
  const unsigned short* Vg = VT + (size_t)b * O_ * S_;
  const float scale = 0.04419417382415922f;   // 1/sqrt(512)

  for (int c = 0; c < 64; ++c) {              // 32 keys per chunk
    const int kbase = c * 32;
    const unsigned short* Krow0 = Kg + (size_t)(kbase + m) * O_;
    const unsigned short* Krow1 = Kg + (size_t)(kbase + 16 + m) * O_;

    // ---- two 16x16 score tiles over the 512-wide dot dimension ----
    v8f sacc0 = (v8f){0,0,0,0,0,0,0,0};
    v8f sacc1 = (v8f){0,0,0,0,0,0,0,0};
    ABu qa[2], kb0[2], kb1[2];
    auto loadQK = [&](int t, int s) {
      qa[s].q[0] = *(const u32x4*)&Qs[g][m * 512 + t * 32 + half * 16];
      qa[s].q[1] = *(const u32x4*)&Qs[g][m * 512 + t * 32 + half * 16 + 8];
      const u32x4* k0 = (const u32x4*)(Krow0 + t * 32 + half * 16);
      kb0[s].q[0] = k0[0];
      kb0[s].q[1] = k0[1];
      const u32x4* k1 = (const u32x4*)(Krow1 + t * 32 + half * 16);
      kb1[s].q[0] = k1[0];
      kb1[s].q[1] = k1[1];
    };
    loadQK(0, 0);
#pragma unroll
    for (int t = 0; t < 16; ++t) {
      const int cur = t & 1;
      if (t < 15) loadQK(t + 1, cur ^ 1);    // prefetch next Q/K fragments
      __builtin_amdgcn_sched_barrier(0);
      sacc0 = __builtin_amdgcn_wmma_f32_16x16x32_bf16(
          false, qa[cur].v, false, kb0[cur].v, (short)0, sacc0, false, false);
      sacc1 = __builtin_amdgcn_wmma_f32_16x16x32_bf16(
          false, qa[cur].v, false, kb1[cur].v, (short)0, sacc1, false, false);
      __builtin_amdgcn_sched_barrier(0);
    }

    // ---- online softmax over all 32 columns of this chunk ----
    float p0[8], p1[8], mloc[8];
#pragma unroll
    for (int r = 0; r < 8; ++r) {
      p0[r] = sacc0[r] * scale;
      p1[r] = sacc1[r] * scale;
      mloc[r] = fmaxf(p0[r], p1[r]);
    }
#pragma unroll
    for (int d = 1; d < 16; d <<= 1)
#pragma unroll
      for (int r = 0; r < 8; ++r) mloc[r] = fmaxf(mloc[r], __shfl_xor(mloc[r], d, 32));
    float corr[8];
#pragma unroll
    for (int r = 0; r < 8; ++r) {
      float mn = fmaxf(rmax[r], mloc[r]);
      corr[r] = __expf(rmax[r] - mn);
      rmax[r] = mn;
      p0[r] = __expf(p0[r] - mn);
      p1[r] = __expf(p1[r] - mn);
    }
    float ps[8];
#pragma unroll
    for (int r = 0; r < 8; ++r) ps[r] = p0[r] + p1[r];
#pragma unroll
    for (int d = 1; d < 16; d <<= 1)
#pragma unroll
      for (int r = 0; r < 8; ++r) ps[r] += __shfl_xor(ps[r], d, 32);
#pragma unroll
    for (int r = 0; r < 8; ++r) rsum[r] = rsum[r] * corr[r] + ps[r];
#pragma unroll
    for (int ot = 0; ot < 16; ++ot)
#pragma unroll
      for (int r = 0; r < 8; ++r) acc[ot][r] *= corr[r];

    // ---- stage P (C layout) -> LDS row-major [16 q x 32 k] bf16 ----
#pragma unroll
    for (int r = 0; r < 8; ++r) {
      Ps[wave][(half * 8 + r) * 32 + m]      = f32_to_bf16(p0[r]);
      Ps[wave][(half * 8 + r) * 32 + 16 + m] = f32_to_bf16(p1[r]);
    }

    // ---- out += P(16x32) . V(32 x 256-slice), V-fragments double-buffered ----
    ABu pa;
    pa.q[0] = *(const u32x4*)&Ps[wave][m * 32 + half * 16];
    pa.q[1] = *(const u32x4*)&Ps[wave][m * 32 + half * 16 + 8];
    ABu vb[2];
    auto loadV = [&](int ot, int s) {
      const int o = ohalf * 256 + ot * 16 + m;
      const u32x4* vp = (const u32x4*)(Vg + (size_t)o * S_ + kbase + half * 16);
      vb[s].q[0] = vp[0];
      vb[s].q[1] = vp[1];
    };
    loadV(0, 0);
#pragma unroll
    for (int ot = 0; ot < 16; ++ot) {
      const int cur = ot & 1;
      if (ot < 15) loadV(ot + 1, cur ^ 1);
      __builtin_amdgcn_sched_barrier(0);
      acc[ot] = __builtin_amdgcn_wmma_f32_16x16x32_bf16(
          false, pa.v, false, vb[cur].v, (short)0, acc[ot], false, false);
      __builtin_amdgcn_sched_barrier(0);
    }
  }

  // ---- epilogue: normalize (rcp) and store fp32 ----
  float inv[8];
#pragma unroll
  for (int r = 0; r < 8; ++r) inv[r] = __builtin_amdgcn_rcpf(rsum[r]);
  float* Og = Out + (size_t)(b * S_ + q0) * O_;
#pragma unroll
  for (int ot = 0; ot < 16; ++ot) {
    const int o = ohalf * 256 + ot * 16 + m;
#pragma unroll
    for (int r = 0; r < 8; ++r)
      Og[(size_t)(half * 8 + r) * O_ + o] = acc[ot][r] * inv[r];
  }
}

// ---------------- host side ----------------
extern "C" void kernel_launch(void* const* d_in, const int* in_sizes, int n_in,
                              void* d_out, int out_size, void* d_ws, size_t ws_size,
                              hipStream_t stream) {
  const float* X  = (const float*)d_in[0];
  const float* WQ = (const float*)d_in[1];
  const float* WK = (const float*)d_in[2];
  const float* WV = (const float*)d_in[3];
  float* Out = (float*)d_out;

  char* ws = (char*)d_ws;
  unsigned short* Xbf = (unsigned short*)(ws);                      // 16 MB
  unsigned short* WTq = (unsigned short*)(ws + 16777216);           //  1 MB
  unsigned short* WTk = (unsigned short*)(ws + 16777216 + 1048576);
  unsigned short* WTv = (unsigned short*)(ws + 16777216 + 2097152);
  unsigned short* Qb  = (unsigned short*)(ws + 19922944);           //  8 MB
  unsigned short* Kb  = (unsigned short*)(ws + 19922944 + 8388608);
  unsigned short* VT  = (unsigned short*)(ws + 19922944 + 16777216);

  const int nX8 = (B_ * S_ * D_) / 8;   // 1048576
  cast_x_kernel<<<nX8 / 256, 256, 0, stream>>>(X, Xbf, nX8);

  const int nW = O_ * D_;               // 524288
  transpose_w_kernel<<<nW / 256, 256, 0, stream>>>(WQ, WTq);
  transpose_w_kernel<<<nW / 256, 256, 0, stream>>>(WK, WTk);
  transpose_w_kernel<<<nW / 256, 256, 0, stream>>>(WV, WTv);

  proj_kernel<<<1024, 128, 0, stream>>>(Xbf, WTq, Qb, 0);
  proj_kernel<<<1024, 128, 0, stream>>>(Xbf, WTk, Kb, 0);
  proj_kernel<<<1024, 128, 0, stream>>>(Xbf, WTv, VT, 1);

  attn_kernel<<<256, 128, 0, stream>>>(Qb, Kb, VT, Out);
}
